// BitLinear_89799176225619
// MI455X (gfx1250) — matrix operand order repaired and checked by reading, not compiled, not run on hardware
//
#include <hip/hip_runtime.h>
#include <stdint.h>

typedef _Float16 v16h __attribute__((ext_vector_type(16)));
typedef _Float16 v8h  __attribute__((ext_vector_type(8)));
typedef _Float16 v4h  __attribute__((ext_vector_type(4)));
typedef float    v8f  __attribute__((ext_vector_type(8)));
typedef float    v4f  __attribute__((ext_vector_type(4)));

#define M_DIM 8192
#define N_DIM 4096
#define K_DIM 4096

#define BM 128
#define BN 256
#define BK 32
#define LDA 40          // padded LDS row stride (f16) -> 80B, breaks bank conflicts
#define LDB 40
#define TERN_THRESH 0.33f

// ---------------- CDNA5 async staging primitives (inline asm) ----------------
// GLOBAL_LOAD_ASYNC_TO_LDS_B128: per-lane 16B global -> LDS, tracked by ASYNCcnt.
// VDST VGPR = LDS byte address (low 32 bits of a generic pointer to __shared__).
__device__ __forceinline__ void async_copy_b128(uint32_t lds_addr, const void* gptr) {
  asm volatile("global_load_async_to_lds_b128 %0, %1, off"
               :: "v"(lds_addr), "v"(gptr)
               : "memory");
}
__device__ __forceinline__ void wait_async0() {
  asm volatile("s_wait_asynccnt 0x0" ::: "memory");
}

__device__ __forceinline__ _Float16 ternary_f16(float w) {
  float s = (__builtin_fabsf(w) < TERN_THRESH) ? 0.0f : (w < 0.0f ? -1.0f : 1.0f);
  return (_Float16)s;
}

// W[4096][4096] f32 -> ternary f16, same [n][k] layout (exact in f16)
__global__ __launch_bounds__(256) void ternarize_weights(const float* __restrict__ W,
                                                         _Float16* __restrict__ Wt) {
  size_t i = (size_t)blockIdx.x * 256 + threadIdx.x;   // one float4 per thread
  v4f w = ((const v4f*)W)[i];
  v4h o;
  o[0] = ternary_f16(w[0]); o[1] = ternary_f16(w[1]);
  o[2] = ternary_f16(w[2]); o[3] = ternary_f16(w[3]);
  ((v4h*)Wt)[i] = o;
}

// x f32 -> f16 (only lossy step; products with +/-1 are exact afterwards)
__global__ __launch_bounds__(256) void convert_x_f16(const float* __restrict__ X,
                                                     _Float16* __restrict__ Xh) {
  size_t i = (size_t)blockIdx.x * 256 + threadIdx.x;
  v4f x = ((const v4f*)X)[i];
  v4h o;
  o[0] = (_Float16)x[0]; o[1] = (_Float16)x[1];
  o[2] = (_Float16)x[2]; o[3] = (_Float16)x[3];
  ((v4h*)Xh)[i] = o;
}

// Main GEMM: out[m][n] = sum_k Xh[m][k] * Wt[n][k] + bias[n]
// 256 threads = 8 wave32s (2x4), block tile 128x256, wave tile 64x64,
// K-step 32, double-buffered LDS fed by GLOBAL_LOAD_ASYNC_TO_LDS_B128,
// compute via V_WMMA_F32_16X16X32_F16.
__global__ __launch_bounds__(256) void bitlinear_wmma_f16(
    const _Float16* __restrict__ Xh,
    const _Float16* __restrict__ Wt,
    const float* __restrict__ bias,
    float* __restrict__ out) {
  __shared__ __align__(16) _Float16 As[2][BM * LDA];
  __shared__ __align__(16) _Float16 Bs[2][BN * LDB];

  const int t    = threadIdx.x;
  const int lane = t & 31;                 // wave32
  const int wid  = t >> 5;                 // 0..7
  const int wm   = (wid >> 2) * 64;        // wave M offset in block: 0,64
  const int wn   = (wid & 3)  * 64;        // wave N offset in block: 0..192
  const int lmod = lane & 15;
  const int lhi  = lane >> 4;              // 0: lanes 0-15, 1: lanes 16-31

  const int block_n = blockIdx.x * BN;
  const int block_m = blockIdx.y * BM;

  // Staging: thread owns K-column group ko (8 f16 = 16B) of rows r0 + 64*j
  const int ko = (t & 3) * 8;
  const int r0 = t >> 2;                   // 0..63
  const _Float16* gA = Xh + (size_t)(block_m + r0) * K_DIM + ko;
  const _Float16* gB = Wt + (size_t)(block_n + r0) * K_DIM + ko;
  const size_t rstride = (size_t)64 * K_DIM;

  // LDS byte addresses for this thread's async destinations
  const uint32_t sA = (uint32_t)(uintptr_t)(&As[0][0] + r0 * LDA + ko);
  const uint32_t sB = (uint32_t)(uintptr_t)(&Bs[0][0] + r0 * LDB + ko);
  const uint32_t aBuf = BM * LDA * sizeof(_Float16);   // bytes per A buffer
  const uint32_t bBuf = BN * LDB * sizeof(_Float16);   // bytes per B buffer
  const uint32_t rowA = 64 * LDA * sizeof(_Float16);
  const uint32_t rowB = 64 * LDB * sizeof(_Float16);

  v8f acc[4][4];
  v8f zero = {};
#pragma unroll
  for (int i = 0; i < 4; ++i)
#pragma unroll
    for (int j = 0; j < 4; ++j) acc[i][j] = zero;

  // ---- prologue: async-stage K-step 0 into buffer 0
  async_copy_b128(sA,            gA);
  async_copy_b128(sA + rowA,     gA + rstride);
  async_copy_b128(sB,            gB);
  async_copy_b128(sB + rowB,     gB + rstride);
  async_copy_b128(sB + 2 * rowB, gB + 2 * rstride);
  async_copy_b128(sB + 3 * rowB, gB + 3 * rstride);
  wait_async0();
  __syncthreads();

  const int NSTEP = K_DIM / BK;            // 128
  int buf = 0;
  for (int ks = 0; ks < NSTEP; ++ks) {
    const bool more = (ks + 1) < NSTEP;
    if (more) {
      // async-stage next K-step into the spare buffer (freed by last barrier)
      const int kn = (ks + 1) * BK;
      const uint32_t dA = sA + (buf ^ 1) * aBuf;
      const uint32_t dB = sB + (buf ^ 1) * bBuf;
      async_copy_b128(dA,            gA + kn);
      async_copy_b128(dA + rowA,     gA + rstride + kn);
      async_copy_b128(dB,            gB + kn);
      async_copy_b128(dB + rowB,     gB + rstride + kn);
      async_copy_b128(dB + 2 * rowB, gB + 2 * rstride + kn);
      async_copy_b128(dB + 3 * rowB, gB + 3 * rstride + kn);
      if (ks + 2 < NSTEP) {                // WGP-scope prefetch for K-step after next
        __builtin_prefetch(gA + kn + BK, 0, 3);
        __builtin_prefetch(gB + kn + BK, 0, 3);
      }
    }

    // ---- compute current buffer
    const _Float16* a_base = &As[buf][0];
    const _Float16* b_base = &Bs[buf][0];

    // A fragment, 16-bit 16x32 layout:
    // lanes 0-15: M=lane, elems 0-7 = K 0..7,  elems 8-15 = K 16..23
    // lanes16-31: M=lane-16, elems 0-7 = K 8..15, elems 8-15 = K 24..31
    v16h af[4];
#pragma unroll
    for (int mt = 0; mt < 4; ++mt) {
      const _Float16* p = a_base + (wm + mt * 16 + lmod) * LDA + lhi * 8;
      v8h lo = *(const v8h*)(p);
      v8h hi = *(const v8h*)(p + 16);
      v16h a;
      ((v8h*)&a)[0] = lo;
      ((v8h*)&a)[1] = hi;
      af[mt] = a;
    }
    // B fragment, 32x16: lane holds column N=lane&15; lanes 0-15 K 0..7,16..23,
    // lanes 16-31 K 8..15,24..31 (same K order as A)
#pragma unroll
    for (int nt = 0; nt < 4; ++nt) {
      const _Float16* p = b_base + (wn + nt * 16 + lmod) * LDB + lhi * 8;
      v8h lo = *(const v8h*)(p);
      v8h hi = *(const v8h*)(p + 16);
      v16h bfr;
      ((v8h*)&bfr)[0] = lo;
      ((v8h*)&bfr)[1] = hi;
#pragma unroll
      for (int mt = 0; mt < 4; ++mt) {
        acc[mt][nt] = __builtin_amdgcn_wmma_f32_16x16x32_f16(
            false, af[mt], false, bfr, (short)0, acc[mt][nt], false, false);
      }
    }

    // our async writes into the spare buffer must land before the barrier
    wait_async0();
    buf ^= (int)more;
    __syncthreads();
  }

  // ---- epilogue: C/D layout: VGPR r -> M = r (+8 for lanes 16-31), N = lane&15
#pragma unroll
  for (int nt = 0; nt < 4; ++nt) {
    const int ng = block_n + wn + nt * 16 + lmod;
    const float bv = bias[ng];
#pragma unroll
    for (int mt = 0; mt < 4; ++mt) {
      const int mg = block_m + wm + mt * 16 + lhi * 8;
#pragma unroll
      for (int r = 0; r < 8; ++r) {
        out[(size_t)(mg + r) * N_DIM + ng] = acc[mt][nt][r] + bv;
      }
    }
  }
}

extern "C" void kernel_launch(void* const* d_in, const int* in_sizes, int n_in,
                              void* d_out, int out_size, void* d_ws, size_t ws_size,
                              hipStream_t stream) {
  const float* x    = (const float*)d_in[0];   // [4,2048,4096] f32
  const float* W    = (const float*)d_in[1];   // [4096,4096]   f32
  const float* bias = (const float*)d_in[2];   // [4096]        f32
  float* out        = (float*)d_out;           // [4,2048,4096] f32

  _Float16* Wt = (_Float16*)d_ws;                                        // 32 MB
  _Float16* Xh = (_Float16*)((char*)d_ws +
                             (size_t)N_DIM * K_DIM * sizeof(_Float16));  // 64 MB

  // prologues: each thread handles 4 elements
  ternarize_weights<<<(N_DIM * (size_t)K_DIM) / 1024, 256, 0, stream>>>(W, Wt);
  convert_x_f16<<<((size_t)M_DIM * K_DIM) / 1024, 256, 0, stream>>>(x, Xh);

  dim3 grid(N_DIM / BN, M_DIM / BM);           // 16 x 64 workgroups
  bitlinear_wmma_f16<<<grid, 256, 0, stream>>>(Xh, Wt, bias, out);
}
